// qSAC_10359461118444
// MI455X (gfx1250) — 16-bit and 8-bit matrix operand order repaired and checked by reading, compiled, not run on hardware
//
#include <hip/hip_runtime.h>

typedef __attribute__((ext_vector_type(8)))  int       v8i;
typedef __attribute__((ext_vector_type(8)))  float     v8f;
typedef __attribute__((ext_vector_type(16))) _Float16  v16h;
typedef __attribute__((ext_vector_type(2)))  _Float16  v2h;
typedef __attribute__((ext_vector_type(4)))  int       i4;

__global__ void k_f32_to_i8v4(const float* __restrict__ s, signed char* __restrict__ d, int n4) {
    int i = blockIdx.x * blockDim.x + threadIdx.x;
    if (i < n4) {
        float4 v = ((const float4*)s)[i];
        int b0 = ((int)fminf(fmaxf(rintf(v.x), -128.f), 127.f)) & 255;
        int b1 = ((int)fminf(fmaxf(rintf(v.y), -128.f), 127.f)) & 255;
        int b2 = ((int)fminf(fmaxf(rintf(v.z), -128.f), 127.f)) & 255;
        int b3 = ((int)fminf(fmaxf(rintf(v.w), -128.f), 127.f)) & 255;
        ((int*)d)[i] = b0 | (b1 << 8) | (b2 << 16) | (b3 << 24);
    }
}
__global__ void k_f32_to_f16v2(const float* __restrict__ s, _Float16* __restrict__ d, int n2) {
    int i = blockIdx.x * blockDim.x + threadIdx.x;
    if (i < n2) {
        float2 v = ((const float2*)s)[i];
        v2h h; h.x = (_Float16)v.x; h.y = (_Float16)v.y;
        ((v2h*)d)[i] = h;
    }
}

__device__ __forceinline__ unsigned lds_addr_of(const void* p) {
    return (unsigned)(unsigned long long)(__attribute__((address_space(3))) const void*)p;
}
__device__ __forceinline__ void async_copy16(unsigned ldsOff, const signed char* g) {
    asm volatile("global_load_async_to_lds_b128 %0, %1, off"
                 :: "v"(ldsOff), "v"(g) : "memory");
}
__device__ __forceinline__ void wait_async5() {
    asm volatile("s_wait_asynccnt 0x5" ::: "memory");
}
__device__ __forceinline__ void wait_async0() {
    asm volatile("s_wait_asynccnt 0x0" ::: "memory");
}

__device__ __forceinline__ v8i lds_a_frag_i8(const signed char* sA, int mLocal, int lane) {
    const int h = (lane >> 4) & 1;
    const signed char* row = sA + mLocal * 64 + h * 16;
    union { v8i v; i4 q[2]; } u;
    u.q[0] = *(const i4*)(row);
    u.q[1] = *(const i4*)(row + 32);
    return u.v;
}
__device__ __forceinline__ v8i lds_b_frag_i8(const signed char* sB, int nLocal, int lane) {
    const int h = (lane >> 4) & 1;
    const signed char* col = sB + nLocal * 64 + h * 16;
    union { v8i v; i4 q[2]; } u;
    u.q[0] = *(const i4*)(col);
    u.q[1] = *(const i4*)(col + 32);
    return u.v;
}

template <int EPI>
__global__ __launch_bounds__(256) void k_gemm_i8(
    const signed char* __restrict__ Amat, const signed char* __restrict__ Wmat,
    const float* __restrict__ bias, const float* __restrict__ Mq,
    const float* __restrict__ Sq,
    signed char* __restrict__ out8, _Float16* __restrict__ outh,
    int M, int N, int K)
{
    __shared__ __align__(16) signed char sA[3][64 * 64];
    __shared__ __align__(16) signed char sB[3][256 * 64];

    const int tid   = threadIdx.x;
    const int lane  = tid & 31;
    const int wave  = tid >> 5;
    const int waveM = wave >> 2;
    const int waveN = wave & 3;
    const int m0 = blockIdx.x * 64;
    const int n0 = blockIdx.y * 256;
    const int ldRow = tid >> 2;
    const int ldCol = (tid & 3) * 16;

    const size_t aBase    = (size_t)(m0 + ldRow) * K + ldCol;
    const size_t bBase    = (size_t)(n0 + ldRow) * K + ldCol;
    const size_t rowPitch = (size_t)64 * K;

    const unsigned ldsA0 = lds_addr_of(&sA[0][0]) + (unsigned)(ldRow * 64 + ldCol);
    const unsigned ldsB0 = lds_addr_of(&sB[0][0]) + (unsigned)(ldRow * 64 + ldCol);

    const v8i vz = {0, 0, 0, 0, 0, 0, 0, 0};
    v8i acc[2][4];
#pragma unroll
    for (int i = 0; i < 2; ++i)
#pragma unroll
        for (int j = 0; j < 4; ++j) acc[i][j] = vz;

    const int nk = K >> 6;

    auto issue_tile = [&](int buf, int ks) {
        const size_t ao = aBase + (size_t)ks * 64;
        const size_t bo = bBase + (size_t)ks * 64;
        const unsigned la = ldsA0 + (unsigned)buf * (64 * 64);
        const unsigned lb = ldsB0 + (unsigned)buf * (256 * 64);
        async_copy16(la,            &Amat[ao]);
        async_copy16(lb,            &Wmat[bo]);
        async_copy16(lb + 64 * 64,  &Wmat[bo + rowPitch]);
        async_copy16(lb + 128 * 64, &Wmat[bo + 2 * rowPitch]);
        async_copy16(lb + 192 * 64, &Wmat[bo + 3 * rowPitch]);
    };

    issue_tile(0, 0);

    for (int ks = 0; ks < nk; ++ks) {
        const int  cur  = ks % 3;
        const bool more = (ks + 1) < nk;

        if (more) issue_tile((ks + 1) % 3, ks + 1);

        if (more) wait_async5(); else wait_async0();
        __syncthreads();

        v8i af[2], bf[4];
        af[0] = lds_a_frag_i8(sA[cur], waveM * 32 + (lane & 15), lane);
        af[1] = lds_a_frag_i8(sA[cur], waveM * 32 + 16 + (lane & 15), lane);
#pragma unroll
        for (int nt = 0; nt < 4; ++nt)
            bf[nt] = lds_b_frag_i8(sB[cur], waveN * 64 + nt * 16 + (lane & 15), lane);

#pragma unroll
        for (int mt = 0; mt < 2; ++mt)
#pragma unroll
            for (int nt = 0; nt < 4; ++nt)
                acc[mt][nt] = __builtin_amdgcn_wmma_i32_16x16x64_iu8(
                    true, af[mt], true, bf[nt], acc[mt][nt], false, false);
    }

    const float Mv = Mq[0];
    const float sQ = (EPI == 1) ? Sq[0] : 1.0f;
    const int   h  = lane >> 4, lo = lane & 15;
    float bv[4];
#pragma unroll
    for (int nt = 0; nt < 4; ++nt) bv[nt] = bias[n0 + waveN * 64 + nt * 16 + lo];

#pragma unroll
    for (int mt = 0; mt < 2; ++mt)
#pragma unroll
        for (int nt = 0; nt < 4; ++nt)
#pragma unroll
            for (int r = 0; r < 8; ++r) {
                int m = m0 + waveM * 32 + mt * 16 + r + 8 * h;
                int n = n0 + waveN * 64 + nt * 16 + lo;
                float f = rintf(((float)acc[mt][nt][r] + bv[nt]) * Mv);
                f = fmaxf(f, 0.0f);
                if (EPI == 1) {
                    float q = rintf(f / sQ);
                    q = fminf(fmaxf(q, -128.f), 127.f);
                    out8[(size_t)m * N + n] = (signed char)(int)q;
                } else {
                    outh[(size_t)m * N + n] = (_Float16)f;
                }
            }
}

__device__ __forceinline__ v16h lds_a_frag_f16(const _Float16* sA, int mLocal, int lane) {
    const int h = (lane >> 4) & 1;
    const _Float16* row = sA + mLocal * 32 + h * 8;
    union { v16h v; i4 q[2]; } u;
    u.q[0] = *(const i4*)(row);
    u.q[1] = *(const i4*)(row + 16);
    return u.v;
}
__device__ __forceinline__ v16h lds_b_frag_f16(const _Float16* sB, int nLocal, int lane) {
    const int h = (lane >> 4) & 1;
    const _Float16* col = sB + nLocal * 32 + h * 8;
    union { v16h v; i4 q[2]; } u;
    u.q[0] = *(const i4*)(col);
    u.q[1] = *(const i4*)(col + 16);
    return u.v;
}

__global__ __launch_bounds__(256) void k_gemm_f16(
    const _Float16* __restrict__ Amat, const _Float16* __restrict__ Wmat,
    const float* __restrict__ bias, const float* __restrict__ Mq,
    float* __restrict__ out, int M, int N, int K)
{
    __shared__ __align__(16) _Float16 sA[2][64 * 32];
    __shared__ __align__(16) _Float16 sB[2][64 * 32];

    const int tid   = threadIdx.x;
    const int lane  = tid & 31;
    const int wave  = tid >> 5;
    const int waveM = wave >> 1;
    const int waveN = wave & 1;
    const int m0 = blockIdx.x * 64;
    const int ldRow = tid >> 2;
    const int ldCol = (tid & 3) * 8;

    const size_t aBase = (size_t)(m0 + ldRow) * K + ldCol;
    const size_t bBase = (size_t)ldRow * K + ldCol;

    const v8f fz = {0.f, 0.f, 0.f, 0.f, 0.f, 0.f, 0.f, 0.f};
    v8f acc[2];
    acc[0] = fz; acc[1] = fz;

    const int nk = K >> 5;

    i4 ra = *(const i4*)&Amat[aBase];
    i4 rb = *(const i4*)&Wmat[bBase];
    *(i4*)&sA[0][ldRow * 32 + ldCol] = ra;
    *(i4*)&sB[0][ldRow * 32 + ldCol] = rb;
    __syncthreads();

    for (int ks = 0; ks < nk; ++ks) {
        const int  cur  = ks & 1;
        const int  nxt  = cur ^ 1;
        const bool more = (ks + 1) < nk;

        if (more) {
            ra = *(const i4*)&Amat[aBase + (size_t)(ks + 1) * 32];
            rb = *(const i4*)&Wmat[bBase + (size_t)(ks + 1) * 32];
        }
        if (ks + 2 < nk)
            __builtin_prefetch(&Amat[aBase + (size_t)(ks + 2) * 32], 0, 3);

        v16h af  = lds_a_frag_f16(sA[cur], waveM * 16 + (lane & 15), lane);
        v16h bf0 = lds_b_frag_f16(sB[cur], waveN * 32 + (lane & 15), lane);
        v16h bf1 = lds_b_frag_f16(sB[cur], waveN * 32 + 16 + (lane & 15), lane);

        acc[0] = __builtin_amdgcn_wmma_f32_16x16x32_f16(false, af, false, bf0,
                                                        (short)0, acc[0], false, false);
        acc[1] = __builtin_amdgcn_wmma_f32_16x16x32_f16(false, af, false, bf1,
                                                        (short)0, acc[1], false, false);

        if (more) {
            *(i4*)&sA[nxt][ldRow * 32 + ldCol] = ra;
            *(i4*)&sB[nxt][ldRow * 32 + ldCol] = rb;
        }
        __syncthreads();
    }

    const float Mv = Mq[0];
    const int h = lane >> 4, lo = lane & 15;
    float bv[2];
#pragma unroll
    for (int nt = 0; nt < 2; ++nt) bv[nt] = bias[waveN * 32 + nt * 16 + lo];
#pragma unroll
    for (int nt = 0; nt < 2; ++nt)
#pragma unroll
        for (int r = 0; r < 8; ++r) {
            int m = m0 + waveM * 16 + r + 8 * h;
            int n = waveN * 32 + nt * 16 + lo;
            out[(size_t)m * N + n] = rintf((acc[nt][r] + bv[nt]) * Mv);
        }
}

extern "C" void kernel_launch(void* const* d_in, const int* in_sizes, int n_in,
                              void* d_out, int out_size, void* d_ws, size_t ws_size,
                              hipStream_t stream) {
    const int Bm = 8192, Dk = 512, H = 4096, Ah = 64;

    const float* x      = (const float*)d_in[0];
    const float* w1     = (const float*)d_in[1];
    const float* b1     = (const float*)d_in[2];
    const float* w2     = (const float*)d_in[3];
    const float* b2     = (const float*)d_in[4];
    const float* w_prob = (const float*)d_in[7];
    const float* b_prob = (const float*)d_in[8];
    const float* M1     = (const float*)d_in[9];
    const float* M2     = (const float*)d_in[10];
    const float* M_prob = (const float*)d_in[12];
    const float* s2     = (const float*)d_in[13];
    float* out = (float*)d_out;

    char* ws = (char*)d_ws;
    size_t off = 0;
    auto carve = [&](size_t bytes) { void* p = ws + off; off += (bytes + 255) & ~(size_t)255; return p; };
    signed char* x8   = (signed char*)carve((size_t)Bm * Dk);
    signed char* w1q  = (signed char*)carve((size_t)H * Dk);
    signed char* w2q  = (signed char*)carve((size_t)H * H);
    _Float16*    wpq  = (_Float16*)   carve((size_t)Ah * H * 2);
    signed char* act1 = (signed char*)carve((size_t)Bm * H);
    _Float16*    act2 = (_Float16*)   carve((size_t)Bm * H * 2);

    int n;
    n = (Bm * Dk) / 4; k_f32_to_i8v4 <<<(n + 255) / 256, 256, 0, stream>>>(x, x8, n);
    n = (H * Dk) / 4;  k_f32_to_i8v4 <<<(n + 255) / 256, 256, 0, stream>>>(w1, w1q, n);
    n = (H * H) / 4;   k_f32_to_i8v4 <<<(n + 255) / 256, 256, 0, stream>>>(w2, w2q, n);
    n = (Ah * H) / 2;  k_f32_to_f16v2<<<(n + 255) / 256, 256, 0, stream>>>(w_prob, wpq, n);

    dim3 g12(Bm / 64, H / 256);
    k_gemm_i8<1><<<g12, 256, 0, stream>>>(x8, w1q, b1, M1, s2, act1, ((_Float16*)0), Bm, H, Dk);
    k_gemm_i8<2><<<g12, 256, 0, stream>>>(act1, w2q, b2, M2, s2, ((signed char*)0), act2, Bm, H, H);
    dim3 g3(Bm / 64, 1);
    k_gemm_f16<<<g3, 256, 0, stream>>>(act2, wpq, b_prob, M_prob, out, Bm, Ah, H);
}
